// Qwerky7TimeMix_42915313221840
// MI455X (gfx1250) — matrix operand order repaired and checked by reading, not compiled, not run on hardware
//
#include <hip/hip_runtime.h>
#include <math.h>

typedef __bf16 bf16_t;
typedef __bf16 v16bf __attribute__((ext_vector_type(16)));
typedef __bf16 v8bf  __attribute__((ext_vector_type(8)));
typedef float  v8f   __attribute__((ext_vector_type(8)));

#define T_LEN   2048
#define C_DIM   2048
#define ATT_DIM 512
#define NHEAD   32
#define HEADD   64
#define GREP    4
#define FSTRIDE ((size_t)NHEAD * T_LEN * HEADD)   // one scan-field, in elements

// ---------------------------------------------------------------------------
// CDNA5 async global->LDS 16B copy (ASYNCcnt-tracked). LDS offset = low 32
// bits of the generic address (ISA: LDS aperture addr[31:0] = offset).
// ---------------------------------------------------------------------------
__device__ __forceinline__ void async_copy_b128(void* lds_dst, const void* gsrc)
{
    unsigned lds = (unsigned)(uintptr_t)lds_dst;
    asm volatile("global_load_async_to_lds_b128 %0, %1, off"
                 :: "v"(lds), "v"(gsrc) : "memory");
}
__device__ __forceinline__ void wait_async_zero()
{
    asm volatile("s_wait_asynccnt 0x0" ::: "memory");
}

// ---------------------------------------------------------------------------
// Generic bf16 WMMA GEMM:  out[M][N] = act( A[M][K] @ W[K][N] + bias[N] )
// A is [M][K] (K contiguous), W is PRE-TRANSPOSED [K][N] (N contiguous).
// Block tile: 128(M) x 64(N), K-step 32. 256 threads = 8 waves; each wave
// computes a 32x32 macro-tile = 4 x v_wmma_f32_16x16x32_bf16 per K-step.
// Double-buffered async staging (global_load_async_to_lds_b128): iteration i
// prefetches tile i+1 into the other buffer while WMMAs consume tile i; a
// single s_wait_asynccnt 0 + barrier per K-step publishes the prefetch.
// MODE: 0 = f32 out + bias, 1 = f32 out, 2 = bf16 out.
// Requires: M % 128 == 0, K % 32 == 0, N % 32 == 0 (N edge-guarded vs 64).
// ---------------------------------------------------------------------------
template <int ACT, int MODE>
__global__ void __launch_bounds__(256)
gemm_bf16_wmma(const bf16_t* __restrict__ A, const bf16_t* __restrict__ W,
               const float* __restrict__ bias,
               float* __restrict__ outF, bf16_t* __restrict__ outB,
               int M, int N, int K)
{
    __shared__ bf16_t As[2][128][40];   // [m][k], 80B rows (16B aligned)
    __shared__ bf16_t Ws[2][32][72];    // [k][n], 144B rows (16B aligned)

    const int tid  = threadIdx.x;
    const int lane = tid & 31;
    const int wid  = tid >> 5;
    const int m0   = blockIdx.x * 128;
    const int n0   = blockIdx.y * 64;
    const int tm   = (wid & 3) * 32;    // wave M macro-tile
    const int tn   = (wid >> 2) * 32;   // wave N macro-tile

    // staging coordinates
    const int arow = tid >> 2;           // 0..63
    const int acg  = (tid & 3) * 8;      // A k-chunk: 0,8,16,24
    const int wkr  = tid >> 3;           // 0..31 (W k row)
    const int wng  = (tid & 7) * 8;      // W n-chunk: 0..56
    const bool wok = (n0 + wng) < N;     // N multiple of 32 -> chunk all-or-none

    auto stage = [&](int buf, int k0) {
        async_copy_b128(&As[buf][arow][acg],
                        A + (size_t)(m0 + arow) * K + k0 + acg);
        async_copy_b128(&As[buf][64 + arow][acg],
                        A + (size_t)(m0 + 64 + arow) * K + k0 + acg);
        if (wok)
            async_copy_b128(&Ws[buf][wkr][wng],
                            W + (size_t)(k0 + wkr) * N + n0 + wng);
    };

    v8f acc[2][2] = {};

    // prologue: fill buffer 0
    stage(0, 0);
    wait_async_zero();
    __syncthreads();

    for (int k0 = 0; k0 < K; k0 += 32) {
        const int cur = (k0 >> 5) & 1;
        const bool hasNext = (k0 + 32) < K;

        // prefetch next tile into the other buffer (overlaps WMMAs below);
        // safe: that buffer's last readers finished before the previous
        // end-of-iteration barrier.
        if (hasNext) stage(cur ^ 1, k0 + 32);

        // --- fragments: A via two contiguous 16B LDS loads, B via 16 scalar LDS reads ---
        v16bf afrag[2], bfrag[2];
        const int mr  = tm + (lane & 15);
        const int ahi = lane >> 4;           // hi half-wave: K groups +8
        #pragma unroll
        for (int s = 0; s < 2; ++s) {
            union { v16bf v; v8bf h[2]; } ua;
            ua.h[0] = *(const v8bf*)&As[cur][mr + 16 * s][ahi * 8];
            ua.h[1] = *(const v8bf*)&As[cur][mr + 16 * s][16 + ahi * 8];
            afrag[s] = ua.v;
            union { v16bf v; bf16_t e[16]; } ub;   // B col n = lane&15, k in table order (same as A)
            #pragma unroll
            for (int e = 0; e < 16; ++e)
                ub.e[e] = Ws[cur][(e < 8) ? (ahi * 8 + e) : (16 + ahi * 8 + (e - 8))][tn + 16 * s + (lane & 15)];
            bfrag[s] = ub.v;
        }

        #pragma unroll
        for (int i = 0; i < 2; ++i) {
            #pragma unroll
            for (int j = 0; j < 2; ++j) {
                acc[i][j] = __builtin_amdgcn_wmma_f32_16x16x32_bf16(
                    false, afrag[i], false, bfrag[j], (short)0, acc[i][j],
                    false, false);
            }
        }

        // publish the prefetch: own async copies done, then block-wide barrier
        wait_async_zero();
        __syncthreads();
    }

    // --- epilogue; C/D layout: lane n = lane&15, VGPR p -> m = p + 8*(lane>=16)
    const int hi = lane >> 4;
    #pragma unroll
    for (int j = 0; j < 2; ++j) {
        const int ncol = n0 + tn + 16 * j + (lane & 15);
        if (ncol >= N) continue;
        const float bv = (MODE == 0) ? bias[ncol] : 0.0f;
        #pragma unroll
        for (int i = 0; i < 2; ++i) {
            #pragma unroll
            for (int p = 0; p < 8; ++p) {
                const int row = m0 + tm + 16 * i + p + hi * 8;
                float val = acc[i][j][p] + bv;
                if (ACT == 1) val = tanhf(val);
                if (ACT == 2) val = 1.0f / (1.0f + expf(-val));
                const size_t o = (size_t)row * N + ncol;
                if (MODE == 2) outB[o] = (bf16_t)val;
                else           outF[o] = val;
            }
        }
    }
}

// ---------------------------------------------------------------------------
// f32 -> bf16 converters (straight and transposing: src[R][Cc] -> dst[Cc][R])
// ---------------------------------------------------------------------------
__global__ void cvt_f32_bf16(const float* __restrict__ src,
                             bf16_t* __restrict__ dst, int n)
{
    int i = blockIdx.x * blockDim.x + threadIdx.x;
    if (i < n) dst[i] = (bf16_t)src[i];
}

__global__ void cvtT_f32_bf16(const float* __restrict__ src,
                              bf16_t* __restrict__ dst, int R, int Cc)
{
    int i = blockIdx.x * blockDim.x + threadIdx.x;
    if (i < R * Cc) {
        int r = i / Cc, c = i % Cc;
        dst[(size_t)c * R + r] = (bf16_t)src[i];
    }
}

// ---------------------------------------------------------------------------
// Prep: RoPE, head repeat, v-first mix, kk normalize, decay w.
// One block per (t,h); 64 threads = head dim. Writes scan-friendly [h][t][64].
// ---------------------------------------------------------------------------
__global__ void __launch_bounds__(64)
prep_kernel(const float* __restrict__ r_in, const float* __restrict__ k_in,
            const float* __restrict__ v_in, const float* __restrict__ wlora,
            const float* __restrict__ iclr, const float* __restrict__ vmix,
            const float* __restrict__ v_first, const float* __restrict__ cosT,
            const float* __restrict__ sinT, const float* __restrict__ k_k,
            const float* __restrict__ k_a, float* __restrict__ scan)
{
    const int t = blockIdx.x;
    const int h = blockIdx.y;
    const int d = threadIdx.x;
    const int c  = h * HEADD + d;
    const int hk = h / GREP;
    const int ca = hk * HEADD + d;
    const int dp  = (d < 32) ? d + 32 : d - 32;
    const float sg = (d < 32) ? -1.0f : 1.0f;

    const float cs = cosT[t * HEADD + d];
    const float sn = sinT[t * HEADD + d];

    const float rv = r_in[(size_t)t * C_DIM + c];
    const float rp = r_in[(size_t)t * C_DIM + h * HEADD + dp];
    const float r4 = rv * cs + sg * rp * sn;

    const float kv = k_in[(size_t)t * ATT_DIM + ca];
    const float kp = k_in[(size_t)t * ATT_DIM + hk * HEADD + dp];
    const float k4 = kv * cs + sg * kp * sn;

    const float vrep = v_in[(size_t)t * ATT_DIM + ca];
    const float vf   = vrep + (v_first[(size_t)t * C_DIM + c] - vrep)
                             * vmix[(size_t)t * C_DIM + c];

    const float ic  = iclr[(size_t)t * C_DIM + c];
    const float kkr = k4 * k_k[c];

    __shared__ float red[64];
    red[d] = kkr * kkr;
    __syncthreads();
    for (int s = 32; s > 0; s >>= 1) {
        if (d < s) red[d] += red[d + s];
        __syncthreads();
    }
    const float nrm = fmaxf(sqrtf(red[0]), 1e-12f);
    const float kkn = kkr / nrm;

    const float kf = k4 * (1.0f + (ic - 1.0f) * k_a[c]);

    const float wl = wlora[(size_t)t * C_DIM + c];
    const float sp = log1pf(expf(-wl));                 // softplus(-wl)
    const float w  = expf(-expf(-sp - 0.5f));

    const size_t base = ((size_t)h * T_LEN + t) * HEADD + d;
    scan[0 * FSTRIDE + base] = r4;
    scan[1 * FSTRIDE + base] = w;
    scan[2 * FSTRIDE + base] = kf;
    scan[3 * FSTRIDE + base] = vf;
    scan[4 * FSTRIDE + base] = kkn;
    scan[5 * FSTRIDE + base] = kkn * ic;                // ab
}

// ---------------------------------------------------------------------------
// Sequential WKV scan: one block per head, 64 threads, thread i owns state
// row S[i][0..63] in VGPRs. Per-t vectors staged in LDS.
// S_new[i][j] = S[i][j]*w[j] + (S[i]·(-kk)) * ab[j] + v[i]*k[j];  y[i]=S_new[i]·r
// ---------------------------------------------------------------------------
__global__ void __launch_bounds__(64)
scan_kernel(const float* __restrict__ scan, const float* __restrict__ s_in,
            float* __restrict__ y_out, float* __restrict__ s_out)
{
    const int h = blockIdx.x;
    const int i = threadIdx.x;

    float S[64];
    #pragma unroll
    for (int j = 0; j < 64; ++j)
        S[j] = s_in[((size_t)h * 64 + i) * 64 + j];

    __shared__ float lr[64], lw[64], lk[64], lkk[64], lab[64];
    const size_t hb = (size_t)h * T_LEN * HEADD;

    for (int t = 0; t < T_LEN; ++t) {
        const size_t o = hb + (size_t)t * HEADD + i;
        lr[i]  = scan[0 * FSTRIDE + o];
        lw[i]  = scan[1 * FSTRIDE + o];
        lk[i]  = scan[2 * FSTRIDE + o];
        const float vi = scan[3 * FSTRIDE + o];
        lkk[i] = scan[4 * FSTRIDE + o];
        lab[i] = scan[5 * FSTRIDE + o];
        __syncthreads();

        float sa = 0.0f;
        #pragma unroll
        for (int j = 0; j < 64; ++j) sa -= S[j] * lkk[j];

        float y = 0.0f;
        #pragma unroll
        for (int j = 0; j < 64; ++j) {
            S[j] = S[j] * lw[j] + sa * lab[j] + vi * lk[j];
            y += S[j] * lr[j];
        }
        y_out[o] = y;
        __syncthreads();
    }

    #pragma unroll
    for (int j = 0; j < 64; ++j)
        s_out[((size_t)h * 64 + i) * 64 + j] = S[j];
}

// ---------------------------------------------------------------------------
// GroupNorm + bonus + gate; writes bf16 input of the final o_w GEMM.
// ---------------------------------------------------------------------------
__global__ void __launch_bounds__(64)
post_kernel(const float* __restrict__ y_in, const float* __restrict__ scan,
            const float* __restrict__ g, const float* __restrict__ ln_w,
            const float* __restrict__ ln_b, const float* __restrict__ r_k,
            bf16_t* __restrict__ yg)
{
    const int t = blockIdx.x;
    const int h = blockIdx.y;
    const int d = threadIdx.x;
    const int c = h * HEADD + d;
    const size_t o = ((size_t)h * T_LEN + t) * HEADD + d;

    const float y = y_in[o];
    __shared__ float r1[64], r2[64];
    r1[d] = y; r2[d] = y * y;
    __syncthreads();
    for (int s = 32; s > 0; s >>= 1) {
        if (d < s) { r1[d] += r1[d + s]; r2[d] += r2[d + s]; }
        __syncthreads();
    }
    const float mu  = r1[0] * (1.0f / 64.0f);
    const float var = r2[0] * (1.0f / 64.0f) - mu * mu;
    __syncthreads();

    float yn = (y - mu) * rsqrtf(var + 6.4e-4f);   // EPS = 1e-5 * 64
    yn = yn * ln_w[c] + ln_b[c];

    const float r4 = scan[0 * FSTRIDE + o];
    const float kf = scan[2 * FSTRIDE + o];
    const float vf = scan[3 * FSTRIDE + o];
    r1[d] = r4 * kf * r_k[h * HEADD + d];
    __syncthreads();
    for (int s = 32; s > 0; s >>= 1) {
        if (d < s) r1[d] += r1[d + s];
        __syncthreads();
    }
    const float dot = r1[0];

    const float yf = (yn + dot * vf) * g[(size_t)t * C_DIM + c];
    yg[(size_t)t * C_DIM + c] = (bf16_t)yf;
}

// ---------------------------------------------------------------------------
extern "C" void kernel_launch(void* const* d_in, const int* in_sizes, int n_in,
                              void* d_out, int out_size, void* d_ws, size_t ws_size,
                              hipStream_t stream)
{
    (void)in_sizes; (void)n_in; (void)out_size; (void)ws_size;

    const float* x       = (const float*)d_in[0];
    const float* wkv_in  = (const float*)d_in[1];
    const float* v_first = (const float*)d_in[2];
    const float* cosT    = (const float*)d_in[3];
    const float* sinT    = (const float*)d_in[4];
    const float* w0 = (const float*)d_in[5];
    const float* w1 = (const float*)d_in[6];
    const float* w2 = (const float*)d_in[7];
    const float* a0 = (const float*)d_in[8];
    const float* a1 = (const float*)d_in[9];
    const float* a2 = (const float*)d_in[10];
    const float* v0 = (const float*)d_in[11];
    const float* v1 = (const float*)d_in[12];
    const float* v2 = (const float*)d_in[13];
    const float* g1 = (const float*)d_in[14];
    const float* g2 = (const float*)d_in[15];
    const float* k_k = (const float*)d_in[16];
    const float* k_a = (const float*)d_in[17];
    const float* r_k = (const float*)d_in[18];
    const float* q_w = (const float*)d_in[19];
    const float* q_b = (const float*)d_in[20];
    const float* k_w = (const float*)d_in[21];
    const float* k_b = (const float*)d_in[22];
    const float* v_w = (const float*)d_in[23];
    const float* v_b = (const float*)d_in[24];
    const float* o_w = (const float*)d_in[25];
    const float* ln_w = (const float*)d_in[26];
    const float* ln_b = (const float*)d_in[27];

    const int T = T_LEN, C = C_DIM, ATT = ATT_DIM;

    char* ws = (char*)d_ws;
    size_t off = 0;
    auto alloc = [&](size_t bytes) -> void* {
        void* p = ws + off;
        off += (bytes + 255) & ~(size_t)255;
        return p;
    };

    // All GEMM W operands live in [K][N] (N contiguous) layout.
    bf16_t* xb  = (bf16_t*)alloc((size_t)T * C * 2);
    bf16_t* qwb = (bf16_t*)alloc((size_t)C * C * 2);
    bf16_t* kwb = (bf16_t*)alloc((size_t)ATT * C * 2);
    bf16_t* vwb = (bf16_t*)alloc((size_t)ATT * C * 2);
    bf16_t* owb = (bf16_t*)alloc((size_t)C * C * 2);
    bf16_t* w1c = (bf16_t*)alloc((size_t)C * 96 * 2);
    bf16_t* w2c = (bf16_t*)alloc((size_t)C * 96 * 2);
    bf16_t* a1c = (bf16_t*)alloc((size_t)C * 96 * 2);
    bf16_t* a2c = (bf16_t*)alloc((size_t)C * 96 * 2);
    bf16_t* v1c = (bf16_t*)alloc((size_t)C * 64 * 2);
    bf16_t* v2c = (bf16_t*)alloc((size_t)C * 64 * 2);
    bf16_t* g1c = (bf16_t*)alloc((size_t)C * 256 * 2);
    bf16_t* g2c = (bf16_t*)alloc((size_t)C * 256 * 2);
    float* r_f   = (float*)alloc((size_t)T * C * 4);
    float* katt  = (float*)alloc((size_t)T * ATT * 4);
    float* vatt  = (float*)alloc((size_t)T * ATT * 4);
    float* wlora = (float*)alloc((size_t)T * C * 4);
    float* iclr  = (float*)alloc((size_t)T * C * 4);
    float* vmix  = (float*)alloc((size_t)T * C * 4);
    float* g_f   = (float*)alloc((size_t)T * C * 4);
    bf16_t* hw = (bf16_t*)alloc((size_t)T * 96 * 2);
    bf16_t* ha = (bf16_t*)alloc((size_t)T * 96 * 2);
    bf16_t* hv = (bf16_t*)alloc((size_t)T * 64 * 2);
    bf16_t* hg = (bf16_t*)alloc((size_t)T * 256 * 2);
    float* scan = (float*)alloc(6 * FSTRIDE * 4);
    float* y_f  = (float*)alloc(FSTRIDE * 4);
    bf16_t* yg  = (bf16_t*)alloc((size_t)T * C * 2);

    auto cvt = [&](const float* s, bf16_t* dpt, size_t n) {
        cvt_f32_bf16<<<dim3((unsigned)((n + 255) / 256)), dim3(256), 0, stream>>>(
            s, dpt, (int)n);
    };
    auto cvtT = [&](const float* s, bf16_t* dpt, int R, int Cc) {
        size_t n = (size_t)R * Cc;
        cvtT_f32_bf16<<<dim3((unsigned)((n + 255) / 256)), dim3(256), 0, stream>>>(
            s, dpt, R, Cc);
    };
    // MODE: 0 = f32 out + bias, 1 = f32 out, 2 = bf16 out.
    auto gemm = [&](const bf16_t* Ain, const bf16_t* Wk, const float* bias,
                    float* oF, bf16_t* oB, int M, int N, int K,
                    int act, int mode) {
        dim3 grd(M / 128, (N + 63) / 64);
        #define GCALL(A_, M_) gemm_bf16_wmma<A_, M_><<<grd, dim3(256), 0, stream>>>( \
            Ain, Wk, bias, oF, oB, M, N, K)
        if      (act == 0 && mode == 0) GCALL(0, 0);
        else if (act == 0 && mode == 1) GCALL(0, 1);
        else if (act == 0 && mode == 2) GCALL(0, 2);
        else if (act == 1 && mode == 2) GCALL(1, 2);
        else if (act == 2 && mode == 0) GCALL(2, 0);
        else                            GCALL(2, 2);
        #undef GCALL
    };

    // Precision staging; projection weights [N][K] -> [K][N], LoRA weights
    // are already [K][N] and convert straight.
    cvt(x, xb, (size_t)T * C);
    cvtT(q_w, qwb, C, C);
    cvtT(k_w, kwb, ATT, C);
    cvtT(v_w, vwb, ATT, C);
    cvtT(o_w, owb, C, C);
    cvt(w1, w1c, (size_t)C * 96);
    cvt(w2, w2c, (size_t)96 * C);
    cvt(a1, a1c, (size_t)C * 96);
    cvt(a2, a2c, (size_t)96 * C);
    cvt(v1, v1c, (size_t)C * 64);
    cvt(v2, v2c, (size_t)64 * C);
    cvt(g1, g1c, (size_t)C * 256);
    cvt(g2, g2c, (size_t)256 * C);

    // Projections and LoRA chains (all on WMMA)
    gemm(xb, qwb, q_b, r_f, nullptr, T, C, C, 0, 0);
    gemm(xb, kwb, k_b, katt, nullptr, T, ATT, C, 0, 0);
    gemm(xb, vwb, v_b, vatt, nullptr, T, ATT, C, 0, 0);
    gemm(xb, w1c, nullptr, nullptr, hw, T, 96, C, 1, 2);   // tanh -> bf16
    gemm(hw, w2c, w0, wlora, nullptr, T, C, 96, 0, 0);
    gemm(xb, a1c, nullptr, nullptr, ha, T, 96, C, 0, 2);
    gemm(ha, a2c, a0, iclr, nullptr, T, C, 96, 2, 0);      // sigmoid -> iclr
    gemm(xb, v1c, nullptr, nullptr, hv, T, 64, C, 0, 2);
    gemm(hv, v2c, v0, vmix, nullptr, T, C, 64, 2, 0);      // sigmoid -> v mix
    gemm(xb, g1c, nullptr, nullptr, hg, T, 256, C, 2, 2);  // sigmoid -> bf16
    gemm(hg, g2c, nullptr, g_f, nullptr, T, C, 256, 0, 1);

    prep_kernel<<<dim3(T, NHEAD), dim3(64), 0, stream>>>(
        r_f, katt, vatt, wlora, iclr, vmix, v_first, cosT, sinT, k_k, k_a, scan);

    float* out_f   = (float*)d_out;
    float* s_out   = out_f + (size_t)T * C;
    float* vfv_out = s_out + (size_t)NHEAD * HEADD * HEADD;

    scan_kernel<<<dim3(NHEAD), dim3(64), 0, stream>>>(scan, wkv_in, y_f, s_out);
    post_kernel<<<dim3(T, NHEAD), dim3(64), 0, stream>>>(
        y_f, scan, g_f, ln_w, ln_b, r_k, yg);

    gemm(yg, owb, nullptr, out_f, nullptr, T, C, C, 0, 1);

    hipMemcpyAsync(vfv_out, v_first, (size_t)T * C * sizeof(float),
                   hipMemcpyDeviceToDevice, stream);
}
